// TopKMLPSAE_26826365730926
// MI455X (gfx1250) — matrix operand order repaired and checked by reading, not compiled, not run on hardware
//
#include <hip/hip_runtime.h>
#include <math.h>

// ---------------- problem constants ----------------
#define DIMC   768
#define HIDC   24576
#define DENSEC 3072
#define LC     24          // NUM_LAYERS*2
#define BC     64
#define MROWS  (BC * LC)   // 1536
#define KTOP   16
#define NCHUNK 4096        // HID column chunk for GEMM2 + topk merge (6 chunks)

typedef __attribute__((ext_vector_type(16))) __bf16 v16bf;
typedef __attribute__((ext_vector_type(8)))  float  v8f;

union FragU { struct { uint4 lo, hi; } q; v16bf v; };

__device__ __forceinline__ __bf16 f2bf(float f) {
    union { float f; unsigned u; } c; c.f = f;
    unsigned u = c.u + 0x7FFFu + ((c.u >> 16) & 1u);   // round-to-nearest-even
    union { unsigned short s; __bf16 b; } o; o.s = (unsigned short)(u >> 16);
    return o.b;
}

__device__ __forceinline__ float gelu_exact(float x) {
    return 0.5f * x * (1.0f + erff(x * 0.70710678118654752440f));
}

// CDNA5 async global->LDS copy of one 16B chunk per lane (tracked by ASYNCcnt)
__device__ __forceinline__ void async_cp16(const uint4* lds_dst, const void* gsrc) {
    unsigned l = (unsigned)(uintptr_t)lds_dst;          // LDS byte address (low 32 bits)
    asm volatile("global_load_async_to_lds_b128 %0, %1, off"
                 :: "v"(l), "v"(gsrc) : "memory");
}
__device__ __forceinline__ void wait_async0() {
    asm volatile("s_wait_asynccnt 0x0" ::: "memory");
}

// ---------------- conversion kernels ----------------
__global__ __launch_bounds__(256) void cvt_bf16_k(const float* __restrict__ s,
                                                  __bf16* __restrict__ d, int n) {
    int i = blockIdx.x * 256 + threadIdx.x;
    if (i < n) d[i] = f2bf(s[i]);
}

// xb = bf16(x - decoder_b2), bias broadcast over batch: bias idx = i % (L*DIM)
__global__ __launch_bounds__(256) void xprep_k(const float* __restrict__ x,
                                               const float* __restrict__ b2,
                                               __bf16* __restrict__ xb, int n, int mod) {
    int i = blockIdx.x * 256 + threadIdx.x;
    if (i < n) xb[i] = f2bf(x[i] - b2[i % mod]);
}

// ---------------- LDS-staged WMMA bf16 GEMM: C = op(A[MxK] * B[NxK]^T + bias) -----------
// WG = 256 threads = 8 waves as 2(M) x 4(N); WG tile 64x256, wave tile 32x64.
// Per 32-k-step: A(64x32) + B(256x32) bf16 tiles double-buffered in LDS via async copies.
// BMODE 0: bias[col]   BMODE 1: bias[(m % LMOD)*N + col]   (decoder_b2 per-(l,dim))
template<int BMODE, bool DO_GELU, bool OUT_BF16>
__global__ __launch_bounds__(256) void gemm_lds_k(const __bf16* __restrict__ A,
                                                  const __bf16* __restrict__ Bm,
                                                  const float*  __restrict__ bias,
                                                  void* __restrict__ Cout,
                                                  int M, int N, int K, int LMOD) {
    // buffer layout (uint4 = 16B chunks): A tile = 64 rows * 4 chunks = [0,256),
    // B tile = 256 rows * 4 chunks = [256,1280). Row stride = 4 chunks (64B = 32 bf16).
    __shared__ uint4 sh[2][1280];

    const int tid  = threadIdx.x;
    const int lane = tid & 31;
    const int w    = tid >> 5;
    const int blocksN = N >> 8;
    const int bm = blockIdx.x / blocksN, bn = blockIdx.x % blocksN;
    const int wm = w >> 2, wn = w & 3;
    const int half = lane >> 4, lr = lane & 15;

    // staging assignment: thread -> (row, 16B chunk)
    const int sr = tid >> 2, sc = tid & 3;
    const __bf16* agA = A  + (size_t)(bm * 64  + sr) * K + sc * 8;
    const __bf16* agB = Bm + (size_t)(bn * 256 + sr) * K + sc * 8;

    v8f zero = {0.f,0.f,0.f,0.f,0.f,0.f,0.f,0.f};
    v8f acc[2][4];
    #pragma unroll
    for (int mi = 0; mi < 2; ++mi)
        #pragma unroll
        for (int ni = 0; ni < 4; ++ni) acc[mi][ni] = zero;

    const int KT = K >> 5;
    // prologue: stage k-step 0 into buffer 0
    {
        async_cp16(&sh[0][sr * 4 + sc], agA);
        #pragma unroll
        for (int i = 0; i < 4; ++i)
            async_cp16(&sh[0][256 + (sr + 64 * i) * 4 + sc], agB + (size_t)(64 * i) * K);
    }

    const int arow0 = wm * 32 + lr;            // A fragment rows (within WG tile)
    for (int kt = 0; kt < KT; ++kt) {
        const int buf = kt & 1;
        wait_async0();
        __syncthreads();                       // buffer `buf` ready; prev reads drained
        if (kt + 1 < KT) {                     // stage next k-step into other buffer
            const int k0 = (kt + 1) * 32;
            async_cp16(&sh[buf ^ 1][sr * 4 + sc], agA + k0);
            #pragma unroll
            for (int i = 0; i < 4; ++i)
                async_cp16(&sh[buf ^ 1][256 + (sr + 64 * i) * 4 + sc],
                           agB + (size_t)(64 * i) * K + k0);
        }
        // A fragments: two 16B chunks per lane at (half) and (2+half)
        FragU fa[2];
        #pragma unroll
        for (int mi = 0; mi < 2; ++mi) {
            const int r = arow0 + mi * 16;
            fa[mi].q.lo = sh[buf][r * 4 + half];
            fa[mi].q.hi = sh[buf][r * 4 + 2 + half];
        }
        // B fragments: two 16B chunks per lane at (half) and (2+half)
        FragU fb[4];
        #pragma unroll
        for (int ni = 0; ni < 4; ++ni) {
            const int r = wn * 64 + ni * 16 + lr;
            const int idx = 256 + r * 4 + half;
            fb[ni].q.lo = sh[buf][idx];
            fb[ni].q.hi = sh[buf][idx + 2];
        }
        #pragma unroll
        for (int mi = 0; mi < 2; ++mi)
            #pragma unroll
            for (int ni = 0; ni < 4; ++ni)
                acc[mi][ni] = __builtin_amdgcn_wmma_f32_16x16x32_bf16(
                    false, fa[mi].v, false, fb[ni].v, (short)0, acc[mi][ni],
                    false, false);
    }

    // epilogue: C/D layout -> m = j + 8*half, n = lane&15
    #pragma unroll
    for (int ni = 0; ni < 4; ++ni) {
        const int col = bn * 256 + wn * 64 + ni * 16 + lr;
        float cb = (BMODE == 0) ? bias[col] : 0.f;
        #pragma unroll
        for (int mi = 0; mi < 2; ++mi) {
            const int mbase = bm * 64 + wm * 32 + mi * 16 + half * 8;
            #pragma unroll
            for (int j = 0; j < 8; ++j) {
                const int m = mbase + j;
                float v = acc[mi][ni][j];
                if (BMODE == 0) v += cb;
                else            v += bias[(size_t)(m % LMOD) * N + col];
                if (DO_GELU) v = gelu_exact(v);
                if (OUT_BF16) ((__bf16*)Cout)[(size_t)m * N + col] = f2bf(v);
                else          ((float*) Cout)[(size_t)m * N + col] = v;
            }
        }
    }
}

// ---------------- per-row top-16 merge over one HID chunk (all in LDS) ----------------
__global__ __launch_bounds__(256) void topk_merge_k(const float* __restrict__ zchunk,
                                                    int chunk_base, int first,
                                                    float* __restrict__ topv,
                                                    int* __restrict__ topi) {
    __shared__ float zl[NCHUNK];
    __shared__ float rv[256]; __shared__ int rgi[256]; __shared__ int rloc[256];
    __shared__ float curv[KTOP]; __shared__ int curi[KTOP];
    __shared__ float outv[KTOP]; __shared__ int outi[KTOP];

    const int tid = threadIdx.x;
    const int row = blockIdx.x;
    const float* zr = zchunk + (size_t)row * NCHUNK;
    for (int c = tid; c < NCHUNK; c += 256) zl[c] = zr[c];
    if (tid < KTOP) {
        curv[tid] = first ? -3.0e38f : topv[row * KTOP + tid];
        curi[tid] = first ? -1        : topi[row * KTOP + tid];
    }
    __syncthreads();

    for (int k = 0; k < KTOP; ++k) {
        float bv = -3.0e38f; int bgi = 0x7fffffff; int bloc = -1;
        for (int c = tid; c < NCHUNK; c += 256) {
            float v = zl[c]; int gi = chunk_base + c;
            if (v > bv || (v == bv && gi < bgi)) { bv = v; bgi = gi; bloc = c; }
        }
        if (tid < KTOP) {
            int gi = curi[tid];
            if (gi >= 0) {
                float v = curv[tid];
                if (v > bv || (v == bv && gi < bgi)) { bv = v; bgi = gi; bloc = -2 - tid; }
            }
        }
        rv[tid] = bv; rgi[tid] = bgi; rloc[tid] = bloc;
        __syncthreads();
        for (int s = 128; s > 0; s >>= 1) {
            if (tid < s) {
                float v2 = rv[tid + s]; int g2 = rgi[tid + s];
                if (v2 > rv[tid] || (v2 == rv[tid] && g2 < rgi[tid])) {
                    rv[tid] = v2; rgi[tid] = g2; rloc[tid] = rloc[tid + s];
                }
            }
            __syncthreads();
        }
        if (tid == 0) {
            outv[k] = rv[0]; outi[k] = rgi[0];
            if (rloc[0] >= 0)        zl[rloc[0]] = -3.0e38f;
            else if (rloc[0] <= -2) { curv[-2 - rloc[0]] = -3.0e38f; curi[-2 - rloc[0]] = 0x7ffffffe; }
        }
        __syncthreads();
    }
    if (tid < KTOP) {
        topv[row * KTOP + tid] = outv[tid];
        topi[row * KTOP + tid] = outi[tid];
    }
}

// ---------------- sparse decode: d = gelu(z_topk @ decoder_w1^T + b1) -> bf16 ----------------
__global__ __launch_bounds__(256) void sparse_decode_k(const float* __restrict__ topv,
                                                       const int*   __restrict__ topi,
                                                       const float* __restrict__ w1d,  // [DENSE, HID]
                                                       const float* __restrict__ b1d,
                                                       __bf16* __restrict__ db) {
    __shared__ float lv[KTOP]; __shared__ int li[KTOP];
    const int m = blockIdx.y;
    const int j = blockIdx.x * 256 + threadIdx.x;
    if (threadIdx.x < KTOP) {
        lv[threadIdx.x] = fmaxf(topv[m * KTOP + threadIdx.x], 0.f);  // ReLU on kept values
        li[threadIdx.x] = topi[m * KTOP + threadIdx.x];
    }
    __syncthreads();
    float acc = b1d[j];
    #pragma unroll
    for (int t = 0; t < KTOP; ++t)
        acc += lv[t] * w1d[(size_t)j * HIDC + li[t]];
    db[(size_t)m * DENSEC + j] = f2bf(gelu_exact(acc));
}

// ---------------- host launcher ----------------
extern "C" void kernel_launch(void* const* d_in, const int* in_sizes, int n_in,
                              void* d_out, int out_size, void* d_ws, size_t ws_size,
                              hipStream_t stream) {
    (void)in_sizes; (void)n_in; (void)out_size; (void)ws_size;
    const float* x   = (const float*)d_in[0];
    const float* ew1 = (const float*)d_in[1];
    const float* eb1 = (const float*)d_in[2];
    const float* ew2 = (const float*)d_in[3];
    const float* eb2 = (const float*)d_in[4];
    const float* dw1 = (const float*)d_in[5];
    const float* db1 = (const float*)d_in[6];
    const float* dw2 = (const float*)d_in[7];
    const float* db2 = (const float*)d_in[8];
    float* out = (float*)d_out;

    // workspace layout (all sizes 256B-aligned); peak ~81 MB
    char* ws = (char*)d_ws;
    size_t off = 0;
    __bf16* xb    = (__bf16*)(ws + off); off += (size_t)MROWS * DIMC   * 2;   // 2.36 MB
    __bf16* w1b   = (__bf16*)(ws + off); off += (size_t)DENSEC * DIMC  * 2;   // 4.72 MB
    __bf16* hb    = (__bf16*)(ws + off); off += (size_t)MROWS * DENSEC * 2;   // 9.44 MB
    __bf16* w2db  = (__bf16*)(ws + off); off += (size_t)DIMC * DENSEC  * 2;   // 4.72 MB
    __bf16* dbuf  = (__bf16*)(ws + off); off += (size_t)MROWS * DENSEC * 2;   // 9.44 MB
    float*  topv  = (float*) (ws + off); off += (size_t)MROWS * KTOP   * 4;
    int*    topi  = (int*)   (ws + off); off += (size_t)MROWS * KTOP   * 4;
    __bf16* w2c   = (__bf16*)(ws + off); off += (size_t)NCHUNK * DENSEC * 2;  // 25.2 MB
    float*  zc    = (float*) (ws + off); off += (size_t)MROWS * NCHUNK * 4;   // 25.2 MB

    // 1) input prep / weight conversion to bf16
    {
        int n = MROWS * DIMC;
        xprep_k<<<(n + 255) / 256, 256, 0, stream>>>(x, db2, xb, n, LC * DIMC);
        n = DENSEC * DIMC;
        cvt_bf16_k<<<(n + 255) / 256, 256, 0, stream>>>(ew1, w1b, n);
        n = DIMC * DENSEC;
        cvt_bf16_k<<<(n + 255) / 256, 256, 0, stream>>>(dw2, w2db, n);
    }

    // 2) GEMM1: hb = gelu(xb @ ew1^T + eb1)   [1536 x 3072], K=768
    {
        int blocks = (MROWS / 64) * (DENSEC / 256);
        gemm_lds_k<0, true, true><<<blocks, 256, 0, stream>>>(
            xb, w1b, eb1, hb, MROWS, DENSEC, DIMC, LC);
    }

    // 3) GEMM2 (chunked over HID) + running top-16 merge
    for (int c = 0; c < HIDC / NCHUNK; ++c) {
        int n = NCHUNK * DENSEC;
        cvt_bf16_k<<<(n + 255) / 256, 256, 0, stream>>>(
            ew2 + (size_t)c * NCHUNK * DENSEC, w2c, n);
        int blocks = (MROWS / 64) * (NCHUNK / 256);
        gemm_lds_k<0, false, false><<<blocks, 256, 0, stream>>>(
            hb, w2c, eb2 + (size_t)c * NCHUNK, zc, MROWS, NCHUNK, DENSEC, LC);
        topk_merge_k<<<MROWS, 256, 0, stream>>>(zc, c * NCHUNK, c == 0, topv, topi);
    }

    // 4) sparse decode: dbuf = gelu(relu(top16) @ dw1^T + db1)
    {
        dim3 grid(DENSEC / 256, MROWS);
        sparse_decode_k<<<grid, 256, 0, stream>>>(topv, topi, dw1, db1, dbuf);
    }

    // 5) GEMM4: out = dbuf @ dw2^T + decoder_b2[(m%L), :]   [1536 x 768], K=3072
    {
        int blocks = (MROWS / 64) * (DIMC / 256);
        gemm_lds_k<1, false, false><<<blocks, 256, 0, stream>>>(
            dbuf, w2db, db2, out, MROWS, DIMC, DENSEC, LC);
    }
}
